// Encoder_27943057228347
// MI455X (gfx1250) — compile-verified, matrix operand order checked by reading, not run on hardware
//
#include <hip/hip_runtime.h>
#include <hip/hip_bf16.h>

// ---------------------------------------------------------------------------
// CDNA5 (gfx1250) transformer encoder forward, wave32 WMMA.
// Compile-time-specialized GEMMs, b128 fragment traffic, DPP softmax
// reductions (no LDS round-trips in the attention inner loop).
// ---------------------------------------------------------------------------

typedef __attribute__((ext_vector_type(16))) _Float16 v16h;
typedef __attribute__((ext_vector_type(8)))  _Float16 v8h;
typedef __attribute__((ext_vector_type(4)))  _Float16 v4h;
typedef __attribute__((ext_vector_type(8)))  float    v8f;
typedef __attribute__((ext_vector_type(4)))  float    v4f;

#define TDK   64
#define EPSLN 1e-5f

__device__ inline v8f wmma_f16(v16h a, v16h b, v8f c) {
  return __builtin_amdgcn_wmma_f32_16x16x32_f16(false, a, false, b, (short)0, c,
                                                false, false);
}

// Fragment loader (ISA 7.12.2): lane -> row m=l&15; halves K = hi*8..+7 and
// 16+hi*8..+7. Two aligned b128 loads. Serves A tiles and n-major B tiles.
__device__ inline v16h frag16(const _Float16* P, int ld) {
  const int l = threadIdx.x & 31, m = l & 15, hi = l >> 4;
  const _Float16* p = P + (size_t)m * ld + hi * 8;
  v8h lo = *(const v8h*)(p);
  v8h hv = *(const v8h*)(p + 16);
  return __builtin_shufflevector(lo, hv, 0, 1, 2, 3, 4, 5, 6, 7,
                                 8, 9, 10, 11, 12, 13, 14, 15);
}

// ---- DPP16 allreduce within each 16-lane row (no LDS, no dep counters) ----
// Steps pair lanes with complementary partials: xor1, xor2 (quad_perm),
// across-quad (row_half_mirror), across-bank (row_mirror). DPP row ops never
// cross the wave half, so the two row-groups of the C-fragment stay separate.
#define DPP_QUAD_XOR1 0xB1   // quad_perm [1,0,3,2]
#define DPP_QUAD_XOR2 0x4E   // quad_perm [2,3,0,1]
#define DPP_HALF_MIRR 0x141  // row_half_mirror
#define DPP_ROW_MIRR  0x140  // row_mirror

template <int CTRL>
__device__ inline float dpp_mov(float x) {
  return __int_as_float(__builtin_amdgcn_update_dpp(
      0, __float_as_int(x), CTRL, 0xF, 0xF, true));
}
__device__ inline float rowmax16(float v) {
  v = fmaxf(v, dpp_mov<DPP_QUAD_XOR1>(v));
  v = fmaxf(v, dpp_mov<DPP_QUAD_XOR2>(v));
  v = fmaxf(v, dpp_mov<DPP_HALF_MIRR>(v));
  v = fmaxf(v, dpp_mov<DPP_ROW_MIRR>(v));
  return v;
}
__device__ inline float rowsum16(float v) {
  v += dpp_mov<DPP_QUAD_XOR1>(v);
  v += dpp_mov<DPP_QUAD_XOR2>(v);
  v += dpp_mov<DPP_HALF_MIRR>(v);
  v += dpp_mov<DPP_ROW_MIRR>(v);
  return v;
}

// ---------------------------------------------------------------------------
// 1) embedding * sqrt(D) + sinusoidal PE
// ---------------------------------------------------------------------------
__global__ __launch_bounds__(256) void embed_kernel(
    const int* __restrict__ x, const float* __restrict__ emb,
    float* __restrict__ h, int S, int D) {
  const int bs  = blockIdx.x;
  const int s   = bs % S;
  const int tok = x[bs];
  const float cdiv = -9.2103403719761836f / (float)D;  // -ln(10000)/D
#pragma unroll 4
  for (int d = threadIdx.x; d < D; d += 256) {
    const int i2    = d & ~1;
    const float div = __expf((float)i2 * cdiv);
    const float ang = (float)s * div;
    const float pe  = (d & 1) ? __cosf(ang) : __sinf(ang);
    h[(size_t)bs * D + d] = emb[(size_t)tok * D + d] * 32.0f + pe;
  }
}

// ---------------------------------------------------------------------------
// 2) WMMA GEMM:  C[M,NN] = act(A[M,KK] @ W[KK,NN] + bias)
//    block tile 128x128, K-tile 32, 8 waves (4x2), wave tile 32x64.
//    OUTMODE: 0 = f32, 1 = f16 row-major, 2 = f16 transposed (packed b128).
// ---------------------------------------------------------------------------
#define BM 128
#define BN 128
#define BK 32
#define LDT (BK + 8)

template <typename TIN, int NN, int KK, int RELU, int OUTMODE>
__global__ __launch_bounds__(256) void gemm_kernel(
    const TIN* __restrict__ A, const float* __restrict__ W,
    const float* __restrict__ bias,
    float* __restrict__ Cf, _Float16* __restrict__ Ch, int M) {
  __shared__ _Float16 As[BM * LDT];
  __shared__ _Float16 Bt[BN * LDT];   // transposed W tile: Bt[n][k]

  const int tid   = threadIdx.x;
  const int wave  = tid >> 5;
  const int waveM = wave >> 1;   // 0..3 (32 rows)
  const int waveN = wave & 1;    // 0..1 (64 cols)
  const int row0  = blockIdx.y * BM;
  const int col0  = blockIdx.x * BN;
  const int l     = tid & 31, hi = l >> 4, ln = l & 15;

  // per-thread staging pointers (advanced by BK per k-step)
  const TIN* ap;
  _Float16*  asp;
  if constexpr (sizeof(TIN) == 4) {            // f32 input: v4f -> v4h
    ap  = A + (size_t)(row0 + (tid >> 3)) * KK + ((tid & 7) << 2);
    asp = As + (tid >> 3) * LDT + ((tid & 7) << 2);
  } else {                                     // f16 input: v8h copy
    ap  = A + (size_t)(row0 + (tid >> 2)) * KK + ((tid & 3) << 3);
    asp = As + (tid >> 2) * LDT + ((tid & 3) << 3);
  }
  const float* wp  = W + (size_t)(tid >> 5) * NN + col0 + ((tid & 31) << 2);
  _Float16*    btp = Bt + ((tid & 31) << 2) * LDT + (tid >> 5);

  // fragment base pointers (constant across k-steps)
  const _Float16* afp[2];
  const _Float16* bfp[4];
#pragma unroll
  for (int mt = 0; mt < 2; ++mt) afp[mt] = &As[(waveM * 32 + mt * 16) * LDT];
#pragma unroll
  for (int nt = 0; nt < 4; ++nt) bfp[nt] = &Bt[(waveN * 64 + nt * 16) * LDT];

  v8f acc[2][4] = {};

#pragma unroll 1
  for (int k0 = 0; k0 < KK; k0 += BK) {
    if constexpr (sizeof(TIN) == 4) {
#pragma unroll
      for (int it = 0; it < 4; ++it) {
        v4f x = *(const v4f*)(ap + (size_t)it * 32 * KK);
        v4h hx;
#pragma unroll
        for (int j = 0; j < 4; ++j) hx[j] = (_Float16)x[j];
        *(v4h*)(asp + it * 32 * LDT) = hx;
      }
    } else {
#pragma unroll
      for (int it = 0; it < 2; ++it)
        *(v8h*)(asp + it * 64 * LDT) = *(const v8h*)(ap + (size_t)it * 64 * KK);
    }
#pragma unroll
    for (int it = 0; it < 4; ++it) {
      v4f x = *(const v4f*)(wp + (size_t)it * 8 * NN);
#pragma unroll
      for (int j = 0; j < 4; ++j) btp[j * LDT + it * 8] = (_Float16)x[j];
    }
    if (k0 + BK < KK) {   // global_prefetch_b8 of next k-tiles
      __builtin_prefetch(ap + BK, 0, 1);
      __builtin_prefetch(wp + (size_t)BK * NN, 0, 1);
    }
    __syncthreads();

    v16h af[2], bf[4];
#pragma unroll
    for (int mt = 0; mt < 2; ++mt) af[mt] = frag16(afp[mt], LDT);
#pragma unroll
    for (int nt = 0; nt < 4; ++nt) bf[nt] = frag16(bfp[nt], LDT);
#pragma unroll
    for (int mt = 0; mt < 2; ++mt)
#pragma unroll
      for (int nt = 0; nt < 4; ++nt)
        acc[mt][nt] = wmma_f16(af[mt], bf[nt], acc[mt][nt]);

    __syncthreads();
    ap += BK;
    wp += (size_t)BK * NN;
  }

  // epilogue: bias (+ReLU); single compile-time output path
  float bv[4];
#pragma unroll
  for (int nt = 0; nt < 4; ++nt) bv[nt] = bias[col0 + waveN * 64 + nt * 16 + ln];

#pragma unroll
  for (int mt = 0; mt < 2; ++mt) {
    const int rowb = row0 + waveM * 32 + mt * 16 + 8 * hi;
#pragma unroll
    for (int nt = 0; nt < 4; ++nt) {
      const int col = col0 + waveN * 64 + nt * 16 + ln;
      if constexpr (OUTMODE == 0) {
        float* Cp = Cf + (size_t)rowb * NN + col;
#pragma unroll
        for (int r = 0; r < 8; ++r) {
          float v = acc[mt][nt][r] + bv[nt];
          if constexpr (RELU) v = fmaxf(v, 0.0f);
          Cp[r * NN] = v;
        }
      } else if constexpr (OUTMODE == 1) {
        _Float16* Cp = Ch + (size_t)rowb * NN + col;
#pragma unroll
        for (int r = 0; r < 8; ++r) {
          float v = acc[mt][nt][r] + bv[nt];
          if constexpr (RELU) v = fmaxf(v, 0.0f);
          Cp[r * NN] = (_Float16)v;
        }
      } else {  // f16 transposed: 8 consecutive rows -> one b128 store
        v8h pk;
#pragma unroll
        for (int r = 0; r < 8; ++r) {
          float v = acc[mt][nt][r] + bv[nt];
          if constexpr (RELU) v = fmaxf(v, 0.0f);
          pk[r] = (_Float16)v;
        }
        *(v8h*)(Ch + (size_t)col * M + rowb) = pk;
      }
    }
  }
}

// ---------------------------------------------------------------------------
// 3) flash attention, one wave per (b, h, 16-query tile); online softmax.
//    q,k f16 token-major; v f16 feature-major (transposed); ctx f16 out.
// ---------------------------------------------------------------------------
__global__ __launch_bounds__(32) void attn_kernel(
    const _Float16* __restrict__ q, const _Float16* __restrict__ k,
    const _Float16* __restrict__ vt, const int* __restrict__ mask,
    _Float16* __restrict__ ctx, int S, int H, int D, int Mtok) {
  const int bh = blockIdx.y;
  const int b  = bh / H;
  const int h  = bh % H;
  const int q0 = blockIdx.x * 16;
  const int l  = threadIdx.x & 31, hi = l >> 4, ln = l & 15;

  const _Float16* Qp = q + ((size_t)b * S + q0) * D + h * TDK;
  const v16h aQ0 = frag16(Qp, D);
  const v16h aQ1 = frag16(Qp + 32, D);

  float m_i[8], l_i[8];
  v8f o_acc[4] = {};
#pragma unroll
  for (int r = 0; r < 8; ++r) { m_i[r] = -1e30f; l_i[r] = 0.0f; }

  __shared__ _Float16 Pbuf[16 * 40];

#pragma unroll 1
  for (int kk = 0; kk < S; kk += 32) {
    v8f s0 = {}, s1 = {};
    {
      const _Float16* Kp = k + ((size_t)b * S + kk) * D + h * TDK;
      s0 = wmma_f16(aQ0, frag16(Kp, D), s0);
      s0 = wmma_f16(aQ1, frag16(Kp + 32, D), s0);
    }
    {
      const _Float16* Kp = k + ((size_t)b * S + kk + 16) * D + h * TDK;
      s1 = wmma_f16(aQ0, frag16(Kp, D), s1);
      s1 = wmma_f16(aQ1, frag16(Kp + 32, D), s1);
    }
    // scale + additive mask (one FMA per element)
    const float madd0 = (mask[(size_t)b * S + kk + ln] == 0) ? -1e9f : 0.0f;
    const float madd1 = (mask[(size_t)b * S + kk + 16 + ln] == 0) ? -1e9f : 0.0f;
#pragma unroll
    for (int r = 0; r < 8; ++r) {
      s0[r] = s0[r] * 0.125f + madd0;
      s1[r] = s1[r] * 0.125f + madd1;
    }
    // online softmax: both key tiles share the query row, so one DPP chain
    // per row handles max and one handles sum.
    float p0[8], p1[8];
#pragma unroll
    for (int r = 0; r < 8; ++r) {
      const float rm   = rowmax16(fmaxf(s0[r], s1[r]));
      const float mnew = fmaxf(m_i[r], rm);
      const float f    = __expf(m_i[r] - mnew);
      m_i[r] = mnew;
      p0[r] = __expf(s0[r] - mnew);
      p1[r] = __expf(s1[r] - mnew);
      const float rs = rowsum16(p0[r] + p1[r]);
      l_i[r] = l_i[r] * f + rs;
#pragma unroll
      for (int dt = 0; dt < 4; ++dt) o_acc[dt][r] *= f;
    }
    // C-layout P -> A-fragment via LDS
    __syncthreads();
#pragma unroll
    for (int r = 0; r < 8; ++r) {
      Pbuf[(8 * hi + r) * 40 + ln]      = (_Float16)p0[r];
      Pbuf[(8 * hi + r) * 40 + 16 + ln] = (_Float16)p1[r];
    }
    __syncthreads();
    const v16h aP = frag16(&Pbuf[0], 40);
#pragma unroll
    for (int dt = 0; dt < 4; ++dt) {
      const _Float16* Vp =
          vt + (size_t)(h * TDK + dt * 16) * Mtok + (size_t)b * S + kk;
      o_acc[dt] = wmma_f16(aP, frag16(Vp, Mtok), o_acc[dt]);
    }
    __syncthreads();
  }
#pragma unroll
  for (int r = 0; r < 8; ++r) {
    const float inv = 1.0f / l_i[r];
    const int row   = q0 + r + 8 * hi;
#pragma unroll
    for (int dt = 0; dt < 4; ++dt) {
      const int col = h * TDK + dt * 16 + ln;
      ctx[((size_t)b * S + row) * D + col] = (_Float16)(o_acc[dt][r] * inv);
    }
  }
}

// ---------------------------------------------------------------------------
// 4) out = LayerNorm(x (+ resid)) * g + b  ; one block per row (D=1024)
// ---------------------------------------------------------------------------
__global__ __launch_bounds__(256) void add_ln_kernel(
    const float* __restrict__ xin, const float* __restrict__ resid,
    const float* __restrict__ g, const float* __restrict__ bta,
    float* __restrict__ out, int D, int useResid) {
  const int row = blockIdx.x;
  const int tid = threadIdx.x;
  const float* xr = xin + (size_t)row * D;
  const float* rr = resid + (size_t)row * D;
  __shared__ float red[256];

  float xv[4];
  float s = 0.0f;
#pragma unroll
  for (int j = 0; j < 4; ++j) {
    const int d = tid + j * 256;
    float vv = xr[d];
    if (useResid) vv += rr[d];
    xv[j] = vv;
    s += vv;
  }
  red[tid] = s;
  __syncthreads();
  for (int off = 128; off > 0; off >>= 1) {
    if (tid < off) red[tid] += red[tid + off];
    __syncthreads();
  }
  const float mu = red[0] / (float)D;
  __syncthreads();

  float vs = 0.0f;
#pragma unroll
  for (int j = 0; j < 4; ++j) {
    const float dlt = xv[j] - mu;
    vs += dlt * dlt;
  }
  red[tid] = vs;
  __syncthreads();
  for (int off = 128; off > 0; off >>= 1) {
    if (tid < off) red[tid] += red[tid + off];
    __syncthreads();
  }
  const float rstd = rsqrtf(red[0] / (float)D + EPSLN);

#pragma unroll
  for (int j = 0; j < 4; ++j) {
    const int d = tid + j * 256;
    out[(size_t)row * D + d] = (xv[j] - mu) * rstd * g[d] + bta[d];
  }
}

// ---------------------------------------------------------------------------
// host orchestration
// ---------------------------------------------------------------------------
extern "C" void kernel_launch(void* const* d_in, const int* in_sizes, int n_in,
                              void* d_out, int out_size, void* d_ws, size_t ws_size,
                              hipStream_t stream) {
  const int Bb = 4, Ss = 2048, Dd = 1024, Hh = 16, Ll = 6, Dff = 4096;
  const int Mrows = Bb * Ss;  // 8192

  const int*   x    = (const int*)d_in[0];
  const int*   mask = (const int*)d_in[1];
  const float* emb  = (const float*)d_in[2];
  const float* wq = (const float*)d_in[3];  const float* bq = (const float*)d_in[4];
  const float* wk = (const float*)d_in[5];  const float* bk = (const float*)d_in[6];
  const float* wv = (const float*)d_in[7];  const float* bv = (const float*)d_in[8];
  const float* wo = (const float*)d_in[9];  const float* bo = (const float*)d_in[10];
  const float* w1 = (const float*)d_in[11]; const float* b1 = (const float*)d_in[12];
  const float* w2 = (const float*)d_in[13]; const float* b2 = (const float*)d_in[14];
  const float* g1 = (const float*)d_in[15]; const float* be1 = (const float*)d_in[16];
  const float* g2 = (const float*)d_in[17]; const float* be2 = (const float*)d_in[18];
  const float* gf = (const float*)d_in[19]; const float* bf = (const float*)d_in[20];

  // ws: h(f32) | tmp(f32) | q16 | k16 | vt16 | cx16 ; ff16 aliases q16..cx16
  const size_t tok = (size_t)Mrows * Dd;
  float*    h    = (float*)d_ws;
  float*    tmp  = h + tok;
  _Float16* q16  = (_Float16*)(tmp + tok);
  _Float16* k16  = q16 + tok;
  _Float16* vt16 = k16 + tok;
  _Float16* cx16 = vt16 + tok;
  _Float16* ff16 = q16;  // Mrows*Dff halves == 4*tok, q/k/v/cx dead by then

  embed_kernel<<<Mrows, 256, 0, stream>>>(x, emb, h, Ss, Dd);

  const dim3 gD(Dd / BN, Mrows / BM);    // (8, 64)
  const dim3 gF(Dff / BN, Mrows / BM);   // (32, 64)
  const dim3 gA(Ss / 16, Bb * Hh);       // (128, 64)

  for (int i = 0; i < Ll; ++i) {
    const float* wqi = wq + (size_t)i * Dd * Dd;   const float* bqi = bq + (size_t)i * Dd;
    const float* wki = wk + (size_t)i * Dd * Dd;   const float* bki = bk + (size_t)i * Dd;
    const float* wvi = wv + (size_t)i * Dd * Dd;   const float* bvi = bv + (size_t)i * Dd;
    const float* woi = wo + (size_t)i * Dd * Dd;   const float* boi = bo + (size_t)i * Dd;
    const float* w1i = w1 + (size_t)i * Dd * Dff;  const float* b1i = b1 + (size_t)i * Dff;
    const float* w2i = w2 + (size_t)i * Dff * Dd;  const float* b2i = b2 + (size_t)i * Dd;

    gemm_kernel<float, 1024, 1024, 0, 1><<<gD, 256, 0, stream>>>(
        h, wqi, bqi, nullptr, q16, Mrows);
    gemm_kernel<float, 1024, 1024, 0, 1><<<gD, 256, 0, stream>>>(
        h, wki, bki, nullptr, k16, Mrows);
    gemm_kernel<float, 1024, 1024, 0, 2><<<gD, 256, 0, stream>>>(
        h, wvi, bvi, nullptr, vt16, Mrows);

    attn_kernel<<<gA, 32, 0, stream>>>(q16, k16, vt16, mask, cx16,
                                       Ss, Hh, Dd, Mrows);

    gemm_kernel<_Float16, 1024, 1024, 0, 0><<<gD, 256, 0, stream>>>(
        cx16, woi, boi, tmp, nullptr, Mrows);
    add_ln_kernel<<<Mrows, 256, 0, stream>>>(tmp, h, g1 + (size_t)i * Dd,
                                             be1 + (size_t)i * Dd, h, Dd, 1);

    gemm_kernel<float, 4096, 1024, 1, 1><<<gF, 256, 0, stream>>>(
        h, w1i, b1i, nullptr, ff16, Mrows);
    gemm_kernel<_Float16, 1024, 4096, 0, 0><<<gD, 256, 0, stream>>>(
        ff16, w2i, b2i, tmp, nullptr, Mrows);
    add_ln_kernel<<<Mrows, 256, 0, stream>>>(tmp, h, g2 + (size_t)i * Dd,
                                             be2 + (size_t)i * Dd, h, Dd, 1);
  }

  add_ln_kernel<<<Mrows, 256, 0, stream>>>(h, h, gf, bf, (float*)d_out, Dd, 0);
}
